// MoDBlock_18356690223155
// MI455X (gfx1250) — matrix operand order repaired and checked by reading, not compiled, not run on hardware
//
#include <hip/hip_runtime.h>
#include <hip/hip_bf16.h>

#define B_  4
#define S_  4096
#define D_  2048
#define F_  8192
#define CAP 2048

#define BM 256
#define BN 128
#define KT 32
#define LDT 40   // padded LDS row stride in halves (80B, multiple of 16B)

typedef __attribute__((ext_vector_type(16))) __bf16 v16bf;
typedef __attribute__((ext_vector_type(8)))  __bf16 v8bf;
typedef __attribute__((ext_vector_type(8)))  float  v8f;
typedef int v4i_gcc __attribute__((vector_size(16)));

union Frag { v16bf v; v8bf h[2]; };

#if __has_builtin(__builtin_amdgcn_global_load_async_to_lds_b128)
#define USE_ASYNC 1
#endif

#if __has_builtin(__builtin_amdgcn_s_wait_asynccnt)
#define WAIT_ASYNC(n) __builtin_amdgcn_s_wait_asynccnt(n)
#else
#define WAIT_ASYNC(n) asm volatile("s_wait_asynccnt %0" ::"i"(n) : "memory")
#endif

#ifdef USE_ASYNC
__device__ __forceinline__ void async_cp16(const __bf16* g, __bf16* l) {
    __builtin_amdgcn_global_load_async_to_lds_b128(
        (__attribute__((address_space(1))) v4i_gcc*)(void*)g,
        (__attribute__((address_space(3))) v4i_gcc*)(void*)l,
        0, 0);
}
#endif

__device__ __forceinline__ float gelu_tanh(float x) {
    const float c = 0.7978845608028654f;
    float x3 = x * x * x;
    float t = tanhf(c * (x + 0.044715f * x3));
    return 0.5f * x * (1.0f + t);
}

// ---------------- Router: logits + sigmoid probs, one wave per token ----------
__global__ __launch_bounds__(256) void router_kernel(const float* __restrict__ hidden,
                                                     const float* __restrict__ rw,
                                                     const float* __restrict__ rb,
                                                     float* __restrict__ logits,
                                                     float* __restrict__ probs) {
    int token = blockIdx.x * 8 + (threadIdx.x >> 5);
    int lane  = threadIdx.x & 31;
    const float* h = hidden + (size_t)token * D_;
    float s = 0.f;
    for (int k = lane; k < D_; k += 32) s += h[k] * rw[k];
    for (int off = 16; off > 0; off >>= 1) s += __shfl_down(s, off, 32);
    if (lane == 0) {
        float l = s + rb[0];
        logits[token] = l;
        probs[token]  = 1.0f / (1.0f + __expf(-l));
    }
}

// ---------------- Top-k per batch: bitonic sort in LDS ------------------------
__global__ __launch_bounds__(1024) void topk_kernel(const float* __restrict__ logits,
                                                    const float* __restrict__ probs,
                                                    int* __restrict__ idx_out,
                                                    float* __restrict__ gprob_out) {
    __shared__ float key[S_];
    __shared__ int   val[S_];
    __shared__ int   sel[CAP];
    int b = blockIdx.x;
    const float* L = logits + (size_t)b * S_;
    for (int i = threadIdx.x; i < S_; i += 1024) { key[i] = L[i]; val[i] = i; }
    __syncthreads();
    // sort descending by (logit, -idx)
    for (int k = 2; k <= S_; k <<= 1) {
        for (int j = k >> 1; j > 0; j >>= 1) {
            for (int i = threadIdx.x; i < S_; i += 1024) {
                int ixj = i ^ j;
                if (ixj > i) {
                    float a = key[i], c = key[ixj];
                    int   ia = val[i], ic = val[ixj];
                    bool aGreater = (a > c) || (a == c && ia < ic);
                    bool desc = ((i & k) == 0);
                    bool doSwap = desc ? !aGreater : aGreater;
                    if (doSwap) { key[i] = c; key[ixj] = a; val[i] = ic; val[ixj] = ia; }
                }
            }
            __syncthreads();
        }
    }
    for (int i = threadIdx.x; i < CAP; i += 1024) sel[i] = val[i];
    __syncthreads();
    // sort the selected indices ascending
    for (int k = 2; k <= CAP; k <<= 1) {
        for (int j = k >> 1; j > 0; j >>= 1) {
            for (int i = threadIdx.x; i < CAP; i += 1024) {
                int ixj = i ^ j;
                if (ixj > i) {
                    int a = sel[i], c = sel[ixj];
                    bool asc = ((i & k) == 0);
                    bool doSwap = asc ? (a > c) : (a < c);
                    if (doSwap) { sel[i] = c; sel[ixj] = a; }
                }
            }
            __syncthreads();
        }
    }
    for (int i = threadIdx.x; i < CAP; i += 1024) {
        int t = sel[i];
        idx_out[(size_t)b * CAP + i]   = t;
        gprob_out[(size_t)b * CAP + i] = probs[(size_t)b * S_ + t];
    }
}

// ---------------- Transpose + f32->bf16 convert: dst[c][r] = src[r][c] --------
__global__ __launch_bounds__(256) void transpose_convert_kernel(const float* __restrict__ src,
                                                                __bf16* __restrict__ dst,
                                                                int rows, int cols) {
    __shared__ float tile[32][33];
    int c0 = blockIdx.x * 32, r0 = blockIdx.y * 32;
    int tx = threadIdx.x & 31, ty = threadIdx.x >> 5;
    for (int i = 0; i < 32; i += 8)
        tile[ty + i][tx] = src[(size_t)(r0 + ty + i) * cols + (c0 + tx)];
    __syncthreads();
    for (int i = 0; i < 32; i += 8)
        dst[(size_t)(c0 + ty + i) * rows + (r0 + tx)] = (__bf16)tile[tx][ty + i];
}

// ---------------- Gather selected rows, convert to bf16 -----------------------
__global__ __launch_bounds__(256) void gather_convert_kernel(const float* __restrict__ hidden,
                                                             const int* __restrict__ idx,
                                                             __bf16* __restrict__ Xg) {
    int row = blockIdx.x;          // b*CAP + c
    int b = row / CAP;
    int t = idx[row];
    const float* src = hidden + ((size_t)b * S_ + t) * D_;
    __bf16* dst = Xg + (size_t)row * D_;
    for (int k = threadIdx.x; k < D_; k += blockDim.x) dst[k] = (__bf16)src[k];
}

// ---------------- Pass-through copy: out = hidden -----------------------------
__global__ __launch_bounds__(256) void copy_kernel(const float4* __restrict__ src,
                                                   float4* __restrict__ dst, long n4) {
    long i = (long)blockIdx.x * blockDim.x + threadIdx.x;
    if (i < n4) dst[i] = src[i];
}

// ---------------- Shared WMMA main loop: 256x128 block tile, 8 waves ----------
// Each wave computes a 64x64 sub-tile = 4x4 WMMA accumulators.
// A: [BM x K] bf16 row-major (block base), Bt: [BN x K] bf16 row-major.
// As/Bs are double-buffered LDS arrays: As[2*BM*LDT], Bs[2*BN*LDT].
__device__ __forceinline__ void gemm_mainloop(const __bf16* __restrict__ A,
                                              const __bf16* __restrict__ Bt,
                                              int K, __bf16* As, __bf16* Bs,
                                              v8f acc[4][4]) {
    const int tid  = threadIdx.x;
    const int lane = tid & 31;
    const int wave = tid >> 5;       // 0..7
    const int wm   = wave & 3;       // 4 waves along M (64 rows each)
    const int wn   = wave >> 2;      // 2 waves along N (64 cols each)
    const int srow = tid >> 1;       // staging: row 0..127
    const int scol = (tid & 1) * 16; // halves 0 or 16
    const int hlf  = lane >> 4;      // half-wave id
    const int l16  = lane & 15;

    const int ASTRIDE = BM * LDT;
    const int BSTRIDE = BN * LDT;

#ifdef USE_ASYNC
    // prologue: fill buffer 0 (3 async B128 per thread)
    async_cp16(&A[(size_t)srow * K + scol],         &As[srow * LDT + scol]);
    async_cp16(&A[(size_t)(srow + 128) * K + scol], &As[(srow + 128) * LDT + scol]);
    async_cp16(&Bt[(size_t)srow * K + scol],        &Bs[srow * LDT + scol]);
#endif

    int buf = 0;
    for (int k0 = 0; k0 < K; k0 += KT, buf ^= 1) {
        __bf16* Ab = As + buf * ASTRIDE;
        __bf16* Bb = Bs + buf * BSTRIDE;
#ifdef USE_ASYNC
        if (k0 + KT < K) {
            __bf16* An = As + (buf ^ 1) * ASTRIDE;
            __bf16* Bn = Bs + (buf ^ 1) * BSTRIDE;
            int kn = k0 + KT;
            async_cp16(&A[(size_t)srow * K + kn + scol],         &An[srow * LDT + scol]);
            async_cp16(&A[(size_t)(srow + 128) * K + kn + scol], &An[(srow + 128) * LDT + scol]);
            async_cp16(&Bt[(size_t)srow * K + kn + scol],        &Bn[srow * LDT + scol]);
            WAIT_ASYNC(3);   // in-order: first 3 (current buffer) have landed
        } else {
            WAIT_ASYNC(0);
        }
        __syncthreads();
#else
        *(v8bf*)&Ab[srow * LDT + scol]         = *(const v8bf*)&A[(size_t)srow * K + k0 + scol];
        *(v8bf*)&Ab[(srow + 128) * LDT + scol] = *(const v8bf*)&A[(size_t)(srow + 128) * K + k0 + scol];
        *(v8bf*)&Bb[srow * LDT + scol]         = *(const v8bf*)&Bt[(size_t)srow * K + k0 + scol];
        __syncthreads();
#endif
        Frag a[4], bfr[4];
#pragma unroll
        for (int tm = 0; tm < 4; ++tm) {
            int r = wm * 64 + tm * 16 + l16;
            // 16-bit A layout: VGPR0-3 hold K = hlf*8..+7, VGPR4-7 hold K = 16+hlf*8..+7
            a[tm].h[0] = *(const v8bf*)&Ab[r * LDT + hlf * 8];
            a[tm].h[1] = *(const v8bf*)&Ab[r * LDT + hlf * 8 + 16];
        }
#pragma unroll
        for (int tn = 0; tn < 4; ++tn) {
            int n = wn * 64 + tn * 16 + l16;
            // 32x16 B layout: VGPR0-3 hold K = hlf*8..+7, VGPR4-7 hold K = 16+hlf*8..+7 (matches A)
            bfr[tn].h[0] = *(const v8bf*)&Bb[n * LDT + hlf * 8];
            bfr[tn].h[1] = *(const v8bf*)&Bb[n * LDT + hlf * 8 + 16];
        }
#pragma unroll
        for (int tm = 0; tm < 4; ++tm)
#pragma unroll
            for (int tn = 0; tn < 4; ++tn)
                acc[tm][tn] = __builtin_amdgcn_wmma_f32_16x16x32_bf16(
                    false, a[tm].v, false, bfr[tn].v, (short)0, acc[tm][tn], false, false);
        __syncthreads();
    }
}

// ---------------- GEMM1: H = gelu(X @ w1 + b1), bf16 out ----------------------
__global__ __launch_bounds__(256) void gemm1_kernel(const __bf16* __restrict__ Xg,
                                                    const __bf16* __restrict__ w1t,
                                                    const float* __restrict__ b1,
                                                    __bf16* __restrict__ H) {
    __shared__ __bf16 As[2 * BM * LDT];
    __shared__ __bf16 Bs[2 * BN * LDT];
    int b  = blockIdx.z;
    int m0 = blockIdx.y * BM;
    int n0 = blockIdx.x * BN;
    const __bf16* A  = Xg  + ((size_t)b * CAP + m0) * D_;
    const __bf16* Bt = w1t + (size_t)n0 * D_;

    v8f acc[4][4];
    const v8f zero = {0.f, 0.f, 0.f, 0.f, 0.f, 0.f, 0.f, 0.f};
#pragma unroll
    for (int i = 0; i < 4; ++i)
#pragma unroll
        for (int j = 0; j < 4; ++j) acc[i][j] = zero;

    gemm_mainloop(A, Bt, D_, As, Bs, acc);

    int lane = threadIdx.x & 31, wave = threadIdx.x >> 5;
    int wm = wave & 3, wn = wave >> 2;
    int hlf = lane >> 4, l16 = lane & 15;
#pragma unroll
    for (int tm = 0; tm < 4; ++tm)
#pragma unroll
        for (int tn = 0; tn < 4; ++tn)
#pragma unroll
            for (int r = 0; r < 8; ++r) {
                int m = m0 + wm * 64 + tm * 16 + r + hlf * 8;
                int n = n0 + wn * 64 + tn * 16 + l16;
                float x = acc[tm][tn][r] + b1[n];
                H[((size_t)b * CAP + m) * F_ + n] = (__bf16)gelu_tanh(x);
            }
}

// ---------------- GEMM2: out[b, idx[c], :] = hidden + gprob*(H @ w2 + b2) -----
__global__ __launch_bounds__(256) void gemm2_kernel(const __bf16* __restrict__ H,
                                                    const __bf16* __restrict__ w2t,
                                                    const float* __restrict__ b2,
                                                    const float* __restrict__ hidden,
                                                    const int* __restrict__ idx,
                                                    const float* __restrict__ gprob,
                                                    float* __restrict__ out) {
    __shared__ __bf16 As[2 * BM * LDT];
    __shared__ __bf16 Bs[2 * BN * LDT];
    int b  = blockIdx.z;
    int m0 = blockIdx.y * BM;
    int n0 = blockIdx.x * BN;
    const __bf16* A  = H   + ((size_t)b * CAP + m0) * F_;
    const __bf16* Bt = w2t + (size_t)n0 * F_;

    v8f acc[4][4];
    const v8f zero = {0.f, 0.f, 0.f, 0.f, 0.f, 0.f, 0.f, 0.f};
#pragma unroll
    for (int i = 0; i < 4; ++i)
#pragma unroll
        for (int j = 0; j < 4; ++j) acc[i][j] = zero;

    gemm_mainloop(A, Bt, F_, As, Bs, acc);

    int lane = threadIdx.x & 31, wave = threadIdx.x >> 5;
    int wm = wave & 3, wn = wave >> 2;
    int hlf = lane >> 4, l16 = lane & 15;
#pragma unroll
    for (int tm = 0; tm < 4; ++tm)
#pragma unroll
        for (int r = 0; r < 8; ++r) {
            int m = m0 + wm * 64 + tm * 16 + r + hlf * 8;   // c index
            int token = idx[(size_t)b * CAP + m];
            float gp  = gprob[(size_t)b * CAP + m];
            size_t rowbase = ((size_t)b * S_ + token) * D_;
#pragma unroll
            for (int tn = 0; tn < 4; ++tn) {
                int n = n0 + wn * 64 + tn * 16 + l16;
                float x = acc[tm][tn][r] + b2[n];
                out[rowbase + n] = hidden[rowbase + n] + gp * x;
            }
        }
}

extern "C" void kernel_launch(void* const* d_in, const int* in_sizes, int n_in,
                              void* d_out, int out_size, void* d_ws, size_t ws_size,
                              hipStream_t stream) {
    const float* hidden = (const float*)d_in[0];
    const float* rw     = (const float*)d_in[1];
    const float* rb     = (const float*)d_in[2];
    const float* w1     = (const float*)d_in[3];
    const float* b1     = (const float*)d_in[4];
    const float* w2     = (const float*)d_in[5];
    const float* b2     = (const float*)d_in[6];
    float* out = (float*)d_out;

    char* ws = (char*)d_ws;
    size_t off = 0;
    auto alloc = [&](size_t bytes) -> void* {
        void* p = ws + off;
        off += (bytes + 255) & ~(size_t)255;
        return p;
    };
    float*  logits = (float*)alloc((size_t)B_ * S_ * 4);
    float*  probs  = (float*)alloc((size_t)B_ * S_ * 4);
    int*    idx    = (int*)  alloc((size_t)B_ * CAP * 4);
    float*  gprob  = (float*)alloc((size_t)B_ * CAP * 4);
    __bf16* w1t    = (__bf16*)alloc((size_t)D_ * F_ * 2);
    __bf16* w2t    = (__bf16*)alloc((size_t)F_ * D_ * 2);
    __bf16* Xg     = (__bf16*)alloc((size_t)B_ * CAP * D_ * 2);
    __bf16* Hbuf   = (__bf16*)alloc((size_t)B_ * CAP * F_ * 2);

    // 1. weight transpose + bf16 convert
    transpose_convert_kernel<<<dim3(F_ / 32, D_ / 32), 256, 0, stream>>>(w1, w1t, D_, F_);
    transpose_convert_kernel<<<dim3(D_ / 32, F_ / 32), 256, 0, stream>>>(w2, w2t, F_, D_);
    // 2. router
    router_kernel<<<B_ * S_ / 8, 256, 0, stream>>>(hidden, rw, rb, logits, probs);
    // 3. per-batch top-k selection
    topk_kernel<<<B_, 1024, 0, stream>>>(logits, probs, idx, gprob);
    // 4. gather + convert
    gather_convert_kernel<<<B_ * CAP, 256, 0, stream>>>(hidden, idx, Xg);
    // 5. pass-through copy
    {
        long n4 = (long)B_ * S_ * D_ / 4;
        long blocks = (n4 + 255) / 256;
        copy_kernel<<<(unsigned)blocks, 256, 0, stream>>>((const float4*)hidden, (float4*)out, n4);
    }
    // 6. FFN
    gemm1_kernel<<<dim3(F_ / BN, CAP / BM, B_), 256, 0, stream>>>(Xg, w1t, b1, Hbuf);
    gemm2_kernel<<<dim3(D_ / BN, CAP / BM, B_), 256, 0, stream>>>(Hbuf, w2t, b2, hidden, idx, gprob, out);
}
